// FlowGNNExpert_47304769798449
// MI455X (gfx1250) — matrix operand order repaired and checked by reading, not compiled, not run on hardware
//
#include <hip/hip_runtime.h>
#include <hip/hip_bf16.h>

// ---------- types ----------
typedef __attribute__((ext_vector_type(16))) __bf16 v16bf;
typedef __attribute__((ext_vector_type(8)))  float  v8f;
typedef __attribute__((ext_vector_type(4)))  unsigned int u32x4;

#define HEADS 8
#define DMODEL 128

// ---------- device helpers ----------
__device__ __forceinline__ unsigned short f2bf(float f) {
    unsigned u = __float_as_uint(f);
    unsigned r = (u + 0x7FFFu + ((u >> 16) & 1u)) >> 16;   // round-to-nearest-even
    return (unsigned short)r;
}

__device__ __forceinline__ float gelu_tanh(float x) {
    float x3 = x * x * x;
    return 0.5f * x * (1.0f + tanhf(0.7978845608f * (x + 0.044715f * x3)));
}

// order-preserving float <-> uint encoding for atomicMax-based segment max
__device__ __forceinline__ unsigned enc_ord(float f) {
    int i = __float_as_int(f);
    return (i >= 0) ? ((unsigned)i | 0x80000000u) : ~(unsigned)i;
}
__device__ __forceinline__ float dec_ord(unsigned u) {
    int i = (u & 0x80000000u) ? (int)(u ^ 0x80000000u) : (int)(~u);
    return __int_as_float(i);
}

// ---------- generic f32 -> bf16 cast ----------
__global__ void cast_f32_bf16(const float* __restrict__ in,
                              unsigned short* __restrict__ out, long long n) {
    long long i = (long long)blockIdx.x * blockDim.x + threadIdx.x;
    if (i < n) out[i] = f2bf(in[i]);
}

// ---------- f32 -> bf16 cast + swizzle into WMMA B-fragment order ----------
// Output layout: for matrix `mat`, K-tile kt (32 rows), N-tile j (16 cols), lane l:
//   16 contiguous bf16 = W[k0 + (l>>4)*16 + v][j*16 + (l&15)], v = 0..15
// so each lane's B fragment is a single 32-byte contiguous run.
template <int K, int NC>
__global__ void cast_swz_bf16(const float* __restrict__ in,
                              unsigned short* __restrict__ out, int nmats) {
    constexpr int MS = K * NC;
    constexpr int JT = NC / 16;
    long long idx = (long long)blockIdx.x * blockDim.x + threadIdx.x;
    if (idx >= (long long)nmats * MS) return;
    int mat = (int)(idx / MS);
    int rem = (int)(idx % MS);
    int k = rem / NC, c = rem % NC;
    int kt = k >> 5, r = k & 31;
    int half = r >> 4, v = r & 15;
    int j = c >> 4, ln = c & 15;
    size_t o = (size_t)mat * MS + (((size_t)(kt * JT + j) * 32) + (half * 16 + ln)) * 16 + v;
    out[o] = f2bf(in[idx]);
}

// ---------- LayerNorm over 128-wide rows (one wave32 per row) ----------
__global__ void ln_rows(const float* __restrict__ in,
                        const float* __restrict__ g, const float* __restrict__ b,
                        float* __restrict__ outf, unsigned short* __restrict__ outbf,
                        int n, int gelu_first) {
    int row = blockIdx.x * (blockDim.x >> 5) + (threadIdx.x >> 5);
    if (row >= n) return;
    int lane = threadIdx.x & 31;
    const float* p = in + (size_t)row * DMODEL + lane * 4;
    float4 v = *(const float4*)p;
    if (gelu_first) {
        v.x = gelu_tanh(v.x); v.y = gelu_tanh(v.y);
        v.z = gelu_tanh(v.z); v.w = gelu_tanh(v.w);
    }
    float s = v.x + v.y + v.z + v.w;
#pragma unroll
    for (int m = 16; m >= 1; m >>= 1) s += __shfl_xor(s, m, 32);
    float mean = s * (1.0f / 128.0f);
    float dx = v.x - mean, dy = v.y - mean, dz = v.z - mean, dw = v.w - mean;
    float q = dx * dx + dy * dy + dz * dz + dw * dw;
#pragma unroll
    for (int m = 16; m >= 1; m >>= 1) q += __shfl_xor(q, m, 32);
    float inv = rsqrtf(q * (1.0f / 128.0f) + 1e-5f);
    float4 gg = *(const float4*)(g + lane * 4);
    float4 bb = *(const float4*)(b + lane * 4);
    float o0 = dx * inv * gg.x + bb.x;
    float o1 = dy * inv * gg.y + bb.y;
    float o2 = dz * inv * gg.z + bb.z;
    float o3 = dw * inv * gg.w + bb.w;
    if (outf)
        *(float4*)(outf + (size_t)row * DMODEL + lane * 4) = make_float4(o0, o1, o2, o3);
    if (outbf) {
        uint2 pk;
        pk.x = (unsigned)f2bf(o0) | ((unsigned)f2bf(o1) << 16);
        pk.y = (unsigned)f2bf(o2) | ((unsigned)f2bf(o3) << 16);
        *(uint2*)(outbf + (size_t)row * DMODEL + lane * 4) = pk;
    }
}

// ---------- WMMA GEMM: C[nrows x NC] = A_bf16[nrows x K] @ Wswz_bf16[K x NC] + bias ----------
// W is pre-swizzled into fragment order; staged in LDS; each B fragment = 2x ds_load_b128.
template <int K, int NC>
__global__ void gemm_bf16_wmma(const unsigned short* __restrict__ A,
                               const unsigned short* __restrict__ W,
                               const float* __restrict__ bias,
                               float* __restrict__ C, int nrows) {
    __shared__ unsigned short sW[K * NC];
    constexpr int total = K * NC;
    for (int i = threadIdx.x * 8; i < total; i += blockDim.x * 8)
        *(u32x4*)(sW + i) = *(const u32x4*)(W + i);
    __syncthreads();

    int lane = threadIdx.x & 31;
    int tile = blockIdx.x * (blockDim.x >> 5) + (threadIdx.x >> 5);
    int r0 = tile * 16;
    if (r0 >= nrows) return;                  // wave-uniform: EXEC stays all-ones for WMMA
    int half = lane >> 4;                     // K-half selector (contiguous K order, matches B swizzle)
    int ln = lane & 15;

    constexpr int JT = NC / 16, KT = K / 32;
    v8f acc[JT] = {};

    int row = r0 + ln;
    if (row >= nrows) row = nrows - 1;        // clamp keeps EXEC full
    const unsigned short* arow = A + (size_t)row * K;

#pragma unroll
    for (int kt = 0; kt < KT; ++kt) {
        int k0 = kt * 32;
        // A fragment: contiguous K order (lanes0-15 K 0..15, lanes16-31 K 16..31), matching B swizzle
        union { v16bf v; u32x4 q[2]; } au;
        au.q[0] = *(const u32x4*)(arow + k0 + half * 16);
        au.q[1] = *(const u32x4*)(arow + k0 + half * 16 + 8);
#pragma unroll
        for (int j = 0; j < JT; ++j) {
            // B fragment: pre-swizzled, 32 contiguous bytes per lane
            const unsigned short* sb = sW + (((kt * JT + j) * 32) + lane) * 16;
            union { v16bf v; u32x4 q[2]; } bu;
            bu.q[0] = *(const u32x4*)(sb);
            bu.q[1] = *(const u32x4*)(sb + 8);
            acc[j] = __builtin_amdgcn_wmma_f32_16x16x32_bf16(
                false, au.v, false, bu.v, (short)0, acc[j], false, false);
        }
    }
    // C layout: VGPR t -> row r0 + t + 8*half, col = 16*j + ln
#pragma unroll
    for (int j = 0; j < JT; ++j) {
        int col = j * 16 + ln;
        float bv = bias[col];
#pragma unroll
        for (int t = 0; t < 8; ++t) {
            int row2 = r0 + t + half * 8;
            if (row2 < nrows) C[(size_t)row2 * NC + col] = acc[j][t] + bv;
        }
    }
}

// ---------- edge kernels: one thread per (edge, head) ----------
__global__ void edge_logit_kernel(const int* __restrict__ src, const int* __restrict__ dst,
                                  const float* __restrict__ ea,
                                  const float* __restrict__ xl, const float* __restrict__ xr,
                                  const float* __restrict__ We, const float* __restrict__ att,
                                  float* __restrict__ logit, unsigned* __restrict__ amax, int E) {
    long long idx = (long long)blockIdx.x * blockDim.x + threadIdx.x;
    if (idx >= (long long)E * HEADS) return;
    int e = (int)(idx >> 3);
    int hh = (int)(idx & 7);
    int s = src[e], d = dst[e];
    const float* pl = xl + (size_t)s * DMODEL + hh * 16;
    const float* pr = xr + (size_t)d * DMODEL + hh * 16;
    __builtin_prefetch(pl, 0, 3);   // global_prefetch_b8
    __builtin_prefetch(pr, 0, 3);
    float a0 = ea[(size_t)e * 3 + 0];
    float a1 = ea[(size_t)e * 3 + 1];
    float a2 = ea[(size_t)e * 3 + 2];
    float acc = 0.0f;
#pragma unroll
    for (int c = 0; c < 16; ++c) {
        int ch = hh * 16 + c;
        float eec = a0 * We[ch] + a1 * We[DMODEL + ch] + a2 * We[2 * DMODEL + ch];
        float m = pl[c] + pr[c] + eec;
        m = (m > 0.0f) ? m : 0.2f * m;        // leaky_relu
        acc += m * att[ch];
    }
    logit[idx] = acc;
    atomicMax(amax + (size_t)d * HEADS + hh, enc_ord(acc));
}

__global__ void edge_exp_kernel(const int* __restrict__ dst, float* __restrict__ logit,
                                const unsigned* __restrict__ amax, float* __restrict__ den, int E) {
    long long idx = (long long)blockIdx.x * blockDim.x + threadIdx.x;
    if (idx >= (long long)E * HEADS) return;
    int e = (int)(idx >> 3);
    int hh = (int)(idx & 7);
    int d = dst[e];
    float mx = dec_ord(amax[(size_t)d * HEADS + hh]);
    float ex = __expf(logit[idx] - mx);
    logit[idx] = ex;
    atomicAdd(den + (size_t)d * HEADS + hh, ex);
}

__global__ void edge_agg_kernel(const int* __restrict__ src, const int* __restrict__ dst,
                                const float* __restrict__ ex, const float* __restrict__ den,
                                const float* __restrict__ xl, float* __restrict__ agg, int E) {
    long long idx = (long long)blockIdx.x * blockDim.x + threadIdx.x;
    if (idx >= (long long)E * HEADS) return;
    int e = (int)(idx >> 3);
    int hh = (int)(idx & 7);
    int s = src[e], d = dst[e];
    float alpha = ex[idx] / den[(size_t)d * HEADS + hh];
    const float4* pj = (const float4*)(xl + (size_t)s * DMODEL + hh * 16);
    __builtin_prefetch(pj, 0, 3);
    float* pa = agg + (size_t)d * DMODEL + hh * 16;
#pragma unroll
    for (int q2 = 0; q2 < 4; ++q2) {
        float4 v = pj[q2];
        atomicAdd(pa + q2 * 4 + 0, alpha * v.x);
        atomicAdd(pa + q2 * 4 + 1, alpha * v.y);
        atomicAdd(pa + q2 * 4 + 2, alpha * v.z);
        atomicAdd(pa + q2 * 4 + 3, alpha * v.w);
    }
}

// ---------- elementwise ----------
__global__ void delta_kernel(float* __restrict__ agg, const float* __restrict__ conv_b,
                             unsigned short* __restrict__ dbf, long long n) {
    long long i = (long long)blockIdx.x * blockDim.x + threadIdx.x;
    if (i >= n) return;
    int c = (int)(i & (DMODEL - 1));
    float v = gelu_tanh(agg[i] + conv_b[c]);
    agg[i] = v;            // delta (f32) kept in place
    dbf[i] = f2bf(v);      // bf16 copy for gate GEMM input
}

__global__ void hid_act_kernel(const float* __restrict__ ghid,
                               unsigned short* __restrict__ hbf, long long n) {
    long long i = (long long)blockIdx.x * blockDim.x + threadIdx.x;
    if (i >= n) return;
    hbf[i] = f2bf(gelu_tanh(ghid[i]));
}

__global__ void update_kernel(float* __restrict__ h, const float* __restrict__ gout,
                              const float* __restrict__ delta, long long n) {
    long long i = (long long)blockIdx.x * blockDim.x + threadIdx.x;
    if (i >= n) return;
    float g = 1.0f / (1.0f + __expf(-gout[i]));
    h[i] += g * delta[i];
}

// ---------- host ----------
extern "C" void kernel_launch(void* const* d_in, const int* in_sizes, int n_in,
                              void* d_out, int out_size, void* d_ws, size_t ws_size,
                              hipStream_t stream) {
    const float* x       = (const float*)d_in[0];
    const int*   ei      = (const int*)d_in[1];
    const float* ea      = (const float*)d_in[2];
    const float* Win_w   = (const float*)d_in[3];
    const float* Win_b   = (const float*)d_in[4];
    const float* ln_in_g = (const float*)d_in[5];
    const float* ln_in_b = (const float*)d_in[6];
    const float* ln_g    = (const float*)d_in[7];
    const float* ln_b    = (const float*)d_in[8];
    const float* Wl_w    = (const float*)d_in[9];
    const float* Wl_b    = (const float*)d_in[10];
    const float* Wr_w    = (const float*)d_in[11];
    const float* Wr_b    = (const float*)d_in[12];
    const float* We_w    = (const float*)d_in[13];
    const float* att     = (const float*)d_in[14];
    const float* conv_b  = (const float*)d_in[15];
    const float* g1_w    = (const float*)d_in[16];
    const float* g1_b    = (const float*)d_in[17];
    const float* g2_w    = (const float*)d_in[18];
    const float* g2_b    = (const float*)d_in[19];
    const float* fn_g    = (const float*)d_in[20];
    const float* fn_b    = (const float*)d_in[21];

    const int N = in_sizes[0] / DMODEL;
    const int E = in_sizes[1] / 2;
    const int L = in_sizes[7] / DMODEL;
    const int* src = ei;
    const int* dst = ei + E;

    // workspace layout, 256B-aligned sub-buffers
    char* wsb = (char*)d_ws;
    size_t off = 0;
    auto alloc = [&](size_t bytes) -> void* {
        void* p = wsb + off;
        off += (bytes + 255) & ~(size_t)255;
        return p;
    };
    float* h    = (float*)alloc((size_t)N * DMODEL * 4);
    float* xl   = (float*)alloc((size_t)N * DMODEL * 4);
    float* xr   = (float*)alloc((size_t)N * DMODEL * 4);
    float* agg  = (float*)alloc((size_t)N * DMODEL * 4);
    float* gout = (float*)alloc((size_t)N * DMODEL * 4);
    float* exb  = (float*)alloc((size_t)E * HEADS * 4);
    unsigned* amax = (unsigned*)alloc((size_t)N * HEADS * 4);
    float* den  = (float*)alloc((size_t)N * HEADS * 4);
    float* ghid = (float*)alloc((size_t)N * 64 * 4);
    unsigned short* zbf   = (unsigned short*)alloc((size_t)N * DMODEL * 2);
    unsigned short* hbf   = (unsigned short*)alloc((size_t)N * 64 * 2);
    unsigned short* Winbf = (unsigned short*)alloc((size_t)DMODEL * DMODEL * 2);
    unsigned short* Wlbf  = (unsigned short*)alloc((size_t)L * DMODEL * DMODEL * 2);
    unsigned short* Wrbf  = (unsigned short*)alloc((size_t)L * DMODEL * DMODEL * 2);
    unsigned short* g1bf  = (unsigned short*)alloc((size_t)L * DMODEL * 64 * 2);
    unsigned short* g2bf  = (unsigned short*)alloc((size_t)L * 64 * DMODEL * 2);

    auto grid1 = [](long long n) { return dim3((unsigned)((n + 255) / 256)); };
    const dim3 blk(256);

    // bf16 weight conversion + fragment swizzle (once per launch; deterministic)
    cast_swz_bf16<128, 128><<<grid1(16384), blk, 0, stream>>>(Win_w, Winbf, 1);
    cast_swz_bf16<128, 128><<<grid1((long long)L * 16384), blk, 0, stream>>>(Wl_w, Wlbf, L);
    cast_swz_bf16<128, 128><<<grid1((long long)L * 16384), blk, 0, stream>>>(Wr_w, Wrbf, L);
    cast_swz_bf16<128, 64><<<grid1((long long)L * 8192), blk, 0, stream>>>(g1_w, g1bf, L);
    cast_swz_bf16<64, 128><<<grid1((long long)L * 8192), blk, 0, stream>>>(g2_w, g2bf, L);

    const int tiles = (N + 15) / 16;
    const dim3 gemmGrid((tiles + 7) / 8);           // 8 waves per block
    const dim3 rowGrid((N + 7) / 8);                // 8 rows per block (wave per row)
    const long long eh = (long long)E * HEADS;

    // input projection: h = LN(gelu(x @ Win + b))
    cast_f32_bf16<<<grid1((long long)N * DMODEL), blk, 0, stream>>>(x, zbf, (long long)N * DMODEL);
    gemm_bf16_wmma<128, 128><<<gemmGrid, blk, 0, stream>>>(zbf, Winbf, Win_b, xl, N);
    ln_rows<<<rowGrid, blk, 0, stream>>>(xl, ln_in_g, ln_in_b, h, nullptr, N, 1);

    for (int l = 0; l < L; ++l) {
        // pre-norm -> bf16 GEMM input
        ln_rows<<<rowGrid, blk, 0, stream>>>(h, ln_g + l * DMODEL, ln_b + l * DMODEL,
                                             nullptr, zbf, N, 0);
        gemm_bf16_wmma<128, 128><<<gemmGrid, blk, 0, stream>>>(
            zbf, Wlbf + (size_t)l * 16384, Wl_b + l * DMODEL, xl, N);
        gemm_bf16_wmma<128, 128><<<gemmGrid, blk, 0, stream>>>(
            zbf, Wrbf + (size_t)l * 16384, Wr_b + l * DMODEL, xr, N);

        hipMemsetAsync(amax, 0, (size_t)N * HEADS * 4, stream);
        hipMemsetAsync(den, 0, (size_t)N * HEADS * 4, stream);
        hipMemsetAsync(agg, 0, (size_t)N * DMODEL * 4, stream);

        edge_logit_kernel<<<grid1(eh), blk, 0, stream>>>(
            src, dst, ea, xl, xr, We_w + (size_t)l * 3 * DMODEL,
            att + (size_t)l * DMODEL, exb, amax, E);
        edge_exp_kernel<<<grid1(eh), blk, 0, stream>>>(dst, exb, amax, den, E);
        edge_agg_kernel<<<grid1(eh), blk, 0, stream>>>(src, dst, exb, den, xl, agg, E);

        // delta = gelu(agg + conv_b); gate MLP; h += sigmoid(gate) * delta
        delta_kernel<<<grid1((long long)N * DMODEL), blk, 0, stream>>>(
            agg, conv_b + l * DMODEL, zbf, (long long)N * DMODEL);
        gemm_bf16_wmma<128, 64><<<gemmGrid, blk, 0, stream>>>(
            zbf, g1bf + (size_t)l * 8192, g1_b + l * 64, ghid, N);
        hid_act_kernel<<<grid1((long long)N * 64), blk, 0, stream>>>(ghid, hbf, (long long)N * 64);
        gemm_bf16_wmma<64, 128><<<gemmGrid, blk, 0, stream>>>(
            hbf, g2bf + (size_t)l * 8192, g2_b + l * DMODEL, gout, N);
        update_kernel<<<grid1((long long)N * DMODEL), blk, 0, stream>>>(
            h, gout, agg, (long long)N * DMODEL);
    }

    // final LayerNorm -> d_out
    ln_rows<<<rowGrid, blk, 0, stream>>>(h, fn_g, fn_b, (float*)d_out, nullptr, N, 0);
}
